// LocalAttentionBlock_20203526160389
// MI455X (gfx1250) — matrix operand order repaired and checked by reading, not compiled, not run on hardware
//
#include <hip/hip_runtime.h>
#include <hip/hip_bf16.h>
#include <cstdint>
#include <cstddef>

#define SEQL   2048
#define DMODEL 1024
#define NHEADS 16
#define HDIM   64
#define DFF    4096
#define WINH   128   // WINDOW/2

typedef __attribute__((ext_vector_type(16))) __bf16 v16bf;
typedef __attribute__((ext_vector_type(8)))  __bf16 v8bf;
typedef __attribute__((ext_vector_type(8)))  float  v8f;

__device__ __forceinline__ unsigned short f2bf(float f) {
  unsigned int u = __float_as_uint(f);
  u += 0x7FFFu + ((u >> 16) & 1u);          // round-to-nearest-even
  return (unsigned short)(u >> 16);
}

__device__ __forceinline__ int clampi(int v, int lo, int hi) {
  return v < lo ? lo : (v > hi ? hi : v);
}

// Assemble a 16-element bf16 WMMA operand from two 16-byte chunks.
// Pure vector IR: two b128 loads land in 8 consecutive VGPRs, no repacking.
__device__ __forceinline__ v16bf frag16(const unsigned short* p0,
                                        const unsigned short* p1) {
  v8bf lo = *(const v8bf*)p0;
  v8bf hi = *(const v8bf*)p1;
  return __builtin_shufflevector(lo, hi, 0, 1, 2, 3, 4, 5, 6, 7,
                                 8, 9, 10, 11, 12, 13, 14, 15);
}

// ---------------------------------------------------------------- cvt f32->bf16
__global__ void cvt_f32_bf16(const float* __restrict__ in,
                             unsigned short* __restrict__ out, int n) {
  int i = blockIdx.x * blockDim.x + threadIdx.x;
  if (i < n) out[i] = f2bf(in[i]);
}

// ---------------------------------------------------------------- WMMA GEMM
// C[M,N] = act( A_bf16[M,K] @ B_f32[K,N] + bias[N] )
// Block tile 128x64, BK=32, 8 waves, wave tile 32x32 (2x2 WMMA 16x16x32 bf16).
// Double-buffered LDS: global loads of tile t+1 overlap WMMAs of tile t,
// one barrier per K-step.
template <bool RELU, bool OUTBF16>
__global__ __launch_bounds__(256)
void gemm_wmma(const unsigned short* __restrict__ A,
               const float* __restrict__ B,
               const float* __restrict__ bias,
               void* __restrict__ Cout,
               int M, int N, int K) {
  constexpr int BM = 128, BN = 64, BK = 32, LDA = BK + 8, LDB = BK + 8;
  __shared__ __align__(16) unsigned short As[2][BM][LDA];   // [m][k]
  __shared__ __align__(16) unsigned short Bs[2][BN][LDB];   // transposed [n][k]

  const int tid  = threadIdx.x;
  const int lane = tid & 31;
  const int wid  = tid >> 5;
  const int wm   = wid >> 1;     // 0..3
  const int wn   = wid & 1;      // 0..1
  const int lm   = lane & 15;
  const int lh   = lane >> 4;
  const int m0   = blockIdx.y * BM;
  const int n0   = blockIdx.x * BN;

  // per-thread staging coordinates
  const int aRow = tid >> 1;             // A: thread owns 32 contiguous bytes
  const int aCol = (tid & 1) * 16;
  const int bRow = tid >> 3;             // B: thread owns 8 fp32 of one k-row
  const int bCol = (tid & 7) * 8;

  uint4  areg[2];
  float4 breg[2];

  auto loadTile = [&](int k0) {
    const unsigned short* asrc = A + (size_t)(m0 + aRow) * K + k0 + aCol;
    areg[0] = *(const uint4*)asrc;
    areg[1] = *(const uint4*)(asrc + 8);
    const float4* bsrc = (const float4*)(B + (size_t)(k0 + bRow) * N + n0 + bCol);
    breg[0] = bsrc[0];
    breg[1] = bsrc[1];
  };
  auto commitTile = [&](int buf) {
    *(uint4*)&As[buf][aRow][aCol]     = areg[0];
    *(uint4*)&As[buf][aRow][aCol + 8] = areg[1];
    float vals[8] = {breg[0].x, breg[0].y, breg[0].z, breg[0].w,
                     breg[1].x, breg[1].y, breg[1].z, breg[1].w};
#pragma unroll
    for (int j = 0; j < 8; ++j) Bs[buf][bCol + j][bRow] = f2bf(vals[j]);
  };

  v8f acc[2][2] = {};

  loadTile(0);
  commitTile(0);
  __syncthreads();

  const int nk = K / BK;
  for (int t = 0; t < nk; ++t) {
    const int buf = t & 1;
    if (t + 1 < nk) loadTile((t + 1) * BK);   // in flight during WMMAs

    v16bf a[2], b[2];
#pragma unroll
    for (int i = 0; i < 2; ++i) {
      int m = wm * 32 + i * 16 + lm;
      a[i] = frag16(&As[buf][m][lh * 8], &As[buf][m][16 + lh * 8]);
    }
#pragma unroll
    for (int j = 0; j < 2; ++j) {
      int n = wn * 32 + j * 16 + lm;
      b[j] = frag16(&Bs[buf][n][lh * 8], &Bs[buf][n][16 + lh * 8]);
    }
#pragma unroll
    for (int i = 0; i < 2; ++i)
#pragma unroll
      for (int j = 0; j < 2; ++j)
        acc[i][j] = __builtin_amdgcn_wmma_f32_16x16x32_bf16(
            false, a[i], false, b[j], (short)0, acc[i][j], false, false);

    if (t + 1 < nk) commitTile(buf ^ 1);      // other buffer: no conflict
    __syncthreads();
  }

  // ---- epilogue: C layout = lane n, VGPR r -> row r + 8*lh
#pragma unroll
  for (int i = 0; i < 2; ++i)
#pragma unroll
    for (int j = 0; j < 2; ++j) {
      int ng = n0 + wn * 32 + j * 16 + lm;
      float bv = bias[ng];
#pragma unroll
      for (int r = 0; r < 8; ++r) {
        int mg  = m0 + wm * 32 + i * 16 + r + 8 * lh;
        float v = acc[i][j][r] + bv;
        if (RELU) v = fmaxf(v, 0.f);
        if (OUTBF16)
          ((unsigned short*)Cout)[(size_t)mg * N + ng] = f2bf(v);
        else
          ((float*)Cout)[(size_t)mg * N + ng] = v;
      }
    }
}

// ---------------------------------------------------------------- local attention
// 1 wave per (16-query block, head). Flash softmax over the +-128 window,
// 32 keys per step, WMMA bf16 for both QK^T and PV.
__global__ __launch_bounds__(32)
void attn_kernel(const unsigned short* __restrict__ Q,
                 const unsigned short* __restrict__ Kt,
                 const unsigned short* __restrict__ V,
                 unsigned short* __restrict__ O) {
  constexpr int LDP = 40, LDV = 40;
  __shared__ __align__(16) unsigned short Pl[16][LDP];   // P tile 16q x 32k
  __shared__ __align__(16) unsigned short Vt[64][LDV];   // V^T tile [d][key]

  const int lane = threadIdx.x;
  const int lm = lane & 15, lh = lane >> 4;
  const int q0 = blockIdx.x * 16;
  const int h  = blockIdx.y;
  const size_t hoff = (size_t)h * HDIM;

  // Q as two A-fragments (d 0..31, 32..63); rows are d-contiguous in memory
  v16bf qa[2];
#pragma unroll
  for (int f = 0; f < 2; ++f) {
    const unsigned short* base = Q + (size_t)(q0 + lm) * DMODEL + hoff + f * 32;
    qa[f] = frag16(base + lh * 16, base + lh * 16 + 8);
  }

  v8f acc[4] = {};
  float mrun[8], lrun[8];
#pragma unroll
  for (int r = 0; r < 8; ++r) { mrun[r] = -1e30f; lrun[r] = 0.f; }
  const float scale = 0.125f;  // 1/sqrt(64)

  for (int s = 0; s < 9; ++s) {
    int jbase = q0 - WINH + s * 32;        // uniform across wave
    if (jbase >= SEQL) break;
    if (jbase + 32 <= 0) continue;

    __syncthreads();
    // ---- stage V^T: 32 keys x 64 dims.
    // Two key rows per lane; one v_perm_b32 per packed column pair:
    //   even d: bytes {vb.b1, vb.b0, va.b1, va.b0}  sel 0x05040100
    //   odd  d: bytes {vb.b3, vb.b2, va.b3, va.b2}  sel 0x07060302
    {
      int kA = clampi(jbase + 2 * lm, 0, SEQL - 1);
      int kB = clampi(jbase + 2 * lm + 1, 0, SEQL - 1);
      const unsigned short* rA = V + (size_t)kA * DMODEL + hoff + lh * 32;
      const unsigned short* rB = V + (size_t)kB * DMODEL + hoff + lh * 32;
      unsigned int wa[16], wb[16];
#pragma unroll
      for (int c = 0; c < 4; ++c) {
        *(uint4*)&wa[c * 4] = *(const uint4*)(rA + c * 8);
        *(uint4*)&wb[c * 4] = *(const uint4*)(rB + c * 8);
      }
#pragma unroll
      for (int w = 0; w < 16; ++w) {
        unsigned int lo = __builtin_amdgcn_perm(wb[w], wa[w], 0x05040100u);
        unsigned int hi = __builtin_amdgcn_perm(wb[w], wa[w], 0x07060302u);
        *(unsigned int*)&Vt[lh * 32 + 2 * w][2 * lm]     = lo;
        *(unsigned int*)&Vt[lh * 32 + 2 * w + 1][2 * lm] = hi;
      }
    }

    // ---- scores: S[16q x 16k] per key group, contraction over d=64 (2 WMMA)
    v8f sc[2];
#pragma unroll
    for (int kg = 0; kg < 2; ++kg) {
      int key = clampi(jbase + kg * 16 + lm, 0, SEQL - 1);
      const unsigned short* kb = Kt + (size_t)key * DMODEL + hoff + lh * 16;
      v16bf b0 = frag16(kb, kb + 8);            // d = 0..31 slice
      v16bf b1 = frag16(kb + 32, kb + 40);      // d = 32..63 slice
      v8f z = {};
      z = __builtin_amdgcn_wmma_f32_16x16x32_bf16(false, qa[0], false, b0,
                                                  (short)0, z, false, false);
      sc[kg] = __builtin_amdgcn_wmma_f32_16x16x32_bf16(false, qa[1], false, b1,
                                                       (short)0, z, false, false);
    }

    // ---- mask + online softmax (rows live per 16-lane half in C layout)
    float p0v[8], p1v[8], corr[8];
#pragma unroll
    for (int r = 0; r < 8; ++r) {
      int mg = q0 + r + 8 * lh;
      int j0 = jbase + lm;
      int j1 = jbase + 16 + lm;
      int d0 = j0 - mg, d1 = j1 - mg;
      bool v0 = (j0 >= 0) && (j0 < SEQL) && (d0 <= WINH) && (d0 >= -WINH);
      bool v1 = (j1 >= 0) && (j1 < SEQL) && (d1 <= WINH) && (d1 >= -WINH);
      float s0 = v0 ? sc[0][r] * scale : -1e30f;
      float s1 = v1 ? sc[1][r] * scale : -1e30f;
      float t = fmaxf(s0, s1);
      t = fmaxf(t, __shfl_xor(t, 1));
      t = fmaxf(t, __shfl_xor(t, 2));
      t = fmaxf(t, __shfl_xor(t, 4));
      t = fmaxf(t, __shfl_xor(t, 8));
      float mn = fmaxf(mrun[r], t);
      float c  = __expf(mrun[r] - mn);
      float p0 = (s0 <= -1e29f) ? 0.f : __expf(s0 - mn);
      float p1 = (s1 <= -1e29f) ? 0.f : __expf(s1 - mn);
      float rs = p0 + p1;
      rs += __shfl_xor(rs, 1);
      rs += __shfl_xor(rs, 2);
      rs += __shfl_xor(rs, 4);
      rs += __shfl_xor(rs, 8);
      mrun[r] = mn;
      lrun[r] = lrun[r] * c + rs;
      corr[r] = c;
      p0v[r] = p0; p1v[r] = p1;
    }

    // ---- P: C layout -> A-fragment via LDS bounce (bf16)
#pragma unroll
    for (int r = 0; r < 8; ++r) {
      int m = r + 8 * lh;
      Pl[m][lm]      = f2bf(p0v[r]);
      Pl[m][16 + lm] = f2bf(p1v[r]);
    }
    __syncthreads();
    v16bf pa;
    {
      const unsigned short* base = &Pl[lm][0];
      pa = frag16(base + lh * 8, base + 16 + lh * 8);
    }
    // ---- O += P @ V : 4 WMMAs (output d split into 4x16 columns)
#pragma unroll
    for (int nb = 0; nb < 4; ++nb) {
      const unsigned short* base = &Vt[nb * 16 + lm][0];
      v16bf bv = frag16(base + lh * 8, base + 16 + lh * 8);
#pragma unroll
      for (int r = 0; r < 8; ++r) acc[nb][r] *= corr[r];
      acc[nb] = __builtin_amdgcn_wmma_f32_16x16x32_bf16(
          false, pa, false, bv, (short)0, acc[nb], false, false);
    }
  }

  // ---- normalize + store bf16 (interleaved [s][h*64+d] layout for wo GEMM)
#pragma unroll
  for (int r = 0; r < 8; ++r) lrun[r] = 1.f / fmaxf(lrun[r], 1e-20f);
#pragma unroll
  for (int nb = 0; nb < 4; ++nb)
#pragma unroll
    for (int r = 0; r < 8; ++r) {
      int mg  = q0 + r + 8 * lh;
      int col = (int)hoff + nb * 16 + lm;
      O[(size_t)mg * DMODEL + col] = f2bf(acc[nb][r] * lrun[r]);
    }
}

// ---------------------------------------------------------------- LayerNorm
// out = LN(a + b) * g + beta ; optional bf16 mirror for the following GEMM.
__global__ __launch_bounds__(256)
void ln_kernel(const float* __restrict__ a, const float* __restrict__ b,
               const float* __restrict__ g, const float* __restrict__ be,
               float* __restrict__ outf, unsigned short* __restrict__ outb) {
  __shared__ float red[256];
  const int row = blockIdx.x, tid = threadIdx.x;
  float v[4];
  float s = 0.f;
#pragma unroll
  for (int i = 0; i < 4; ++i) {
    int c = tid + i * 256;
    v[i] = a[(size_t)row * DMODEL + c] + b[(size_t)row * DMODEL + c];
    s += v[i];
  }
  red[tid] = s; __syncthreads();
  for (int off = 128; off > 0; off >>= 1) {
    if (tid < off) red[tid] += red[tid + off];
    __syncthreads();
  }
  float mu = red[0] * (1.f / DMODEL);
  __syncthreads();
  float s2 = 0.f;
#pragma unroll
  for (int i = 0; i < 4; ++i) { float d = v[i] - mu; s2 += d * d; }
  red[tid] = s2; __syncthreads();
  for (int off = 128; off > 0; off >>= 1) {
    if (tid < off) red[tid] += red[tid + off];
    __syncthreads();
  }
  float rstd = rsqrtf(red[0] * (1.f / DMODEL) + 1e-5f);
#pragma unroll
  for (int i = 0; i < 4; ++i) {
    int c = tid + i * 256;
    float o = (v[i] - mu) * rstd * g[c] + be[c];
    outf[(size_t)row * DMODEL + c] = o;
    if (outb) outb[(size_t)row * DMODEL + c] = f2bf(o);
  }
}

// ---------------------------------------------------------------- launcher
extern "C" void kernel_launch(void* const* d_in, const int* in_sizes, int n_in,
                              void* d_out, int out_size, void* d_ws, size_t ws_size,
                              hipStream_t stream) {
  (void)in_sizes; (void)n_in; (void)out_size; (void)ws_size;
  const float* x   = (const float*)d_in[0];
  const float* wq  = (const float*)d_in[1];
  const float* bq  = (const float*)d_in[2];
  const float* wk  = (const float*)d_in[3];
  const float* bk  = (const float*)d_in[4];
  const float* wv  = (const float*)d_in[5];
  const float* bv  = (const float*)d_in[6];
  const float* wo  = (const float*)d_in[7];
  const float* bo  = (const float*)d_in[8];
  const float* g1  = (const float*)d_in[9];
  const float* be1 = (const float*)d_in[10];
  const float* wf1 = (const float*)d_in[11];
  const float* bf1 = (const float*)d_in[12];
  const float* wf2 = (const float*)d_in[13];
  const float* bf2 = (const float*)d_in[14];
  const float* g2  = (const float*)d_in[15];
  const float* be2 = (const float*)d_in[16];

  char* ws = (char*)d_ws;
  size_t off = 0;
  auto alloc = [&](size_t bytes) {
    void* p = ws + off;
    off += (bytes + 255) & ~(size_t)255;
    return p;
  };
  const size_t SD = (size_t)SEQL * DMODEL;
  unsigned short* xb    = (unsigned short*)alloc(SD * 2);
  unsigned short* Qb    = (unsigned short*)alloc(SD * 2);
  unsigned short* Kb    = (unsigned short*)alloc(SD * 2);
  unsigned short* Vb    = (unsigned short*)alloc(SD * 2);
  unsigned short* Ob    = (unsigned short*)alloc(SD * 2);
  float*          attno = (float*)alloc(SD * 4);
  float*          hbuf  = (float*)alloc(SD * 4);
  unsigned short* hb    = (unsigned short*)alloc(SD * 2);
  unsigned short* ff1b  = (unsigned short*)alloc((size_t)SEQL * DFF * 2);
  float*          ff2   = (float*)alloc(SD * 4);

  // 1) x -> bf16
  {
    int n = (int)SD;
    cvt_f32_bf16<<<(n + 255) / 256, 256, 0, stream>>>(x, xb, n);
  }
  // 2) Q/K/V projections (bf16 out)
  dim3 gP(DMODEL / 64, SEQL / 128);
  gemm_wmma<false, true><<<gP, 256, 0, stream>>>(xb, wq, bq, Qb, SEQL, DMODEL, DMODEL);
  gemm_wmma<false, true><<<gP, 256, 0, stream>>>(xb, wk, bk, Kb, SEQL, DMODEL, DMODEL);
  gemm_wmma<false, true><<<gP, 256, 0, stream>>>(xb, wv, bv, Vb, SEQL, DMODEL, DMODEL);
  // 3) local attention
  attn_kernel<<<dim3(SEQL / 16, NHEADS), 32, 0, stream>>>(Qb, Kb, Vb, Ob);
  // 4) output projection (fp32 out)
  gemm_wmma<false, false><<<gP, 256, 0, stream>>>(Ob, wo, bo, attno, SEQL, DMODEL, DMODEL);
  // 5) residual + LN1 (fp32 h + bf16 mirror)
  ln_kernel<<<SEQL, 256, 0, stream>>>(x, attno, g1, be1, hbuf, hb);
  // 6) FFN up + ReLU (bf16 out)
  gemm_wmma<true, true><<<dim3(DFF / 64, SEQL / 128), 256, 0, stream>>>(
      hb, wf1, bf1, ff1b, SEQL, DFF, DMODEL);
  // 7) FFN down (fp32 out)
  gemm_wmma<false, false><<<gP, 256, 0, stream>>>(ff1b, wf2, bf2, ff2, SEQL, DMODEL, DFF);
  // 8) residual + LN2 -> d_out
  ln_kernel<<<SEQL, 256, 0, stream>>>(hbuf, ff2, g2, be2, (float*)d_out, nullptr);
}
